// GPT_79173427134481
// MI455X (gfx1250) — matrix operand order repaired and checked by reading, not compiled, not run on hardware
//
#include <hip/hip_runtime.h>
#include <hip/hip_bf16.h>
#include <math.h>

// ---------------------------------------------------------------------------
// GPT forward for MI455X (gfx1250): bf16 WMMA GEMMs with f32 accumulation.
// GEMM panels staged into LDS by the Tensor Data Mover (tensor_load_to_lds),
// double-buffered and synchronized with s_wait_tensorcnt.
// ---------------------------------------------------------------------------

typedef __bf16 bf16_t;
typedef __attribute__((ext_vector_type(16))) __bf16 v16bf;
typedef __attribute__((ext_vector_type(8)))  float  v8f;
typedef uint32_t u32x4 __attribute__((ext_vector_type(4)));
typedef uint32_t u32x8 __attribute__((ext_vector_type(8)));

#define D_MODEL 1024
#define SEQ     1024
#define BATCH   4
#define NHEADS  16
#define HDIM    64
#define HFF     4096
#define VOCAB   32000
#define NTOK    (BATCH * SEQ)   // 4096 rows in the token dimension

__device__ __forceinline__ v8f wmma_bf16(v16bf a, v16bf b, v8f c) {
  // D = A(16x32) * B(32x16) + C, f32 accumulate
  return __builtin_amdgcn_wmma_f32_16x16x32_bf16(
      /*neg_a=*/false, a, /*neg_b=*/false, b,
      /*c_mod=*/(short)0, c, /*reuse_a=*/false, /*reuse_b=*/false);
}

// A-fragment: 16x32 (MxK) 16-bit tile from row-major storage, leading dim ld.
// ISA 7.12.2: lanes 0-15 hold row M=lane, K={0..7,16..23}; lanes 16-31 hold
// row M=lane-16, K={8..15,24..31}.
__device__ __forceinline__ v16bf load_fragA(const bf16_t* p, int ld) {
  const int lane = threadIdx.x & 31;
  const int row  = lane & 15;
  const int kof  = (lane >> 4) << 3;          // 0 or 8
  union { v16bf v; uint4 u[2]; } f;
  const bf16_t* base = p + row * ld + kof;
  f.u[0] = *reinterpret_cast<const uint4*>(base);        // K kof..kof+7
  f.u[1] = *reinterpret_cast<const uint4*>(base + 16);   // K kof+16..kof+23
  return f.v;
}

// B-fragment: 32x16 (KxN) 16-bit tile, sourced from row-major [N,K] storage
// (i.e. B^T) with leading dim ld. Lanes 0-15 hold column N=lane, K={0..7,16..23};
// lanes 16-31 hold column N=lane-16, K={8..15,24..31} (matches A-fragment).
__device__ __forceinline__ v16bf load_fragB(const bf16_t* p, int ld) {
  const int lane = threadIdx.x & 31;
  const int col  = lane & 15;
  const int kof  = (lane >> 4) << 3;          // 0 or 8
  union { v16bf v; uint4 u[2]; } f;
  const bf16_t* base = p + col * ld + kof;
  f.u[0] = *reinterpret_cast<const uint4*>(base);        // K kof..kof+7
  f.u[1] = *reinterpret_cast<const uint4*>(base + 16);   // K kof+16..kof+23
  return f.v;
}

// ---------------------------------------------------------------------------
// Tensor Data Mover: DMA a [rows x 32] bf16 tile (row stride = K elements)
// from global memory into a packed LDS tile (ld = 32). 2D descriptor, so
// VADDR2/VADDR3 are NULL -> 2-operand tensor_load_to_lds form.
// Issued once per wave (caller gates on wave 0); tracked by TENSORcnt.
// ---------------------------------------------------------------------------
__device__ __forceinline__ void tdm_load_panel(uint32_t ldsOff, const bf16_t* gptr,
                                               int K, int rows) {
  const uint64_t ga = (uint64_t)(uintptr_t)gptr;
  u32x4 g0;
  g0[0] = 1u;                                           // count=1, user mode
  g0[1] = ldsOff;                                       // lds_addr (bytes)
  g0[2] = (uint32_t)ga;                                 // global_addr[31:0]
  g0[3] = (uint32_t)((ga >> 32) & 0x1FFFFFFu)           // global_addr[56:32]
        | 0x80000000u;                                  // type=2 ("image")
  u32x8 g1;
  g1[0] = 0x10000u;                                     // data_size=1 -> 2 bytes
  g1[1] = ((uint32_t)K & 0xFFFFu) << 16;                // tensor_dim0[15:0]
  g1[2] = (((uint32_t)K >> 16) & 0xFFFFu)               // tensor_dim0[31:16]
        | ((uint32_t)rows << 16);                       // tensor_dim1[15:0]
  g1[3] = (32u << 16);                                  // tile_dim0 = 32
  g1[4] = (uint32_t)rows;                               // tile_dim1; tile_dim2=0
  g1[5] = (uint32_t)K;                                  // tensor_dim0_stride lo
  g1[6] = 0u;                                           // stride hi / dim1_stride
  g1[7] = 0u;
  asm volatile("tensor_load_to_lds %0, %1" :: "s"(g0), "s"(g1) : "memory");
}

// ---------------------------------------------------------------------------
// f32 -> bf16 conversion
// ---------------------------------------------------------------------------
__global__ __launch_bounds__(256)
void cvt_f32_bf16(const float* __restrict__ in, bf16_t* __restrict__ out, int n) {
  int i = blockIdx.x * 256 + threadIdx.x;
  if (i < n) out[i] = (bf16_t)in[i];
}

// ---------------------------------------------------------------------------
// Embedding * sqrt(D) + sinusoidal positional encoding  ->  h (fp32)
// ---------------------------------------------------------------------------
__global__ __launch_bounds__(256)
void embed_pe_kernel(const int* __restrict__ x, const float* __restrict__ embW,
                     float* __restrict__ h) {
  const int m   = blockIdx.x;                  // token row 0..4095
  const int tid = threadIdx.x;
  const int tok = x[m];
  const float pos = (float)(m & (SEQ - 1));
  const float* ep = embW + (size_t)tok * D_MODEL;
  float* hp = h + (size_t)m * D_MODEL;
  const float kfac = 9.2103403719761836f / (float)D_MODEL;  // ln(10000)/D
#pragma unroll
  for (int i = 0; i < 4; ++i) {
    int d = tid + 256 * i;
    float ang = pos * __expf(-(float)(d & ~1) * kfac);
    float pe  = (d & 1) ? __cosf(ang) : __sinf(ang);
    hp[d] = ep[d] * 32.0f + pe;                // sqrt(1024) = 32
  }
}

// ---------------------------------------------------------------------------
// a = h + LayerNorm(h; g, b), written as bf16. One block per token row.
// ---------------------------------------------------------------------------
__global__ __launch_bounds__(256)
void add_ln_kernel(const float* __restrict__ h, const float* __restrict__ g,
                   const float* __restrict__ be, bf16_t* __restrict__ out) {
  __shared__ float r1[256];
  __shared__ float r2[256];
  const int row = blockIdx.x;
  const int tid = threadIdx.x;
  const float* hp = h + (size_t)row * D_MODEL;
  float v[4];
  float s = 0.f, sq = 0.f;
#pragma unroll
  for (int i = 0; i < 4; ++i) {
    v[i] = hp[tid + 256 * i];
    s  += v[i];
    sq += v[i] * v[i];
  }
  r1[tid] = s; r2[tid] = sq;
  __syncthreads();
  for (int off = 128; off > 0; off >>= 1) {
    if (tid < off) { r1[tid] += r1[tid + off]; r2[tid] += r2[tid + off]; }
    __syncthreads();
  }
  const float mu   = r1[0] * (1.0f / D_MODEL);
  const float var  = r2[0] * (1.0f / D_MODEL) - mu * mu;
  const float rstd = rsqrtf(var + 1e-5f);
  bf16_t* op = out + (size_t)row * D_MODEL;
#pragma unroll
  for (int i = 0; i < 4; ++i) {
    int d = tid + 256 * i;
    float ln = (v[i] - mu) * rstd * g[d] + be[d];
    op[d] = (bf16_t)(v[i] + ln);
  }
}

// ---------------------------------------------------------------------------
// Generic GEMM:  C[M,N] = act(A[M,K] @ W[N,K]^T + bias) (+ residual)
// Block: 128 threads = 4 waves, block tile 128x128, 64x64 per wave,
// 16 v_wmma_f32_16x16x32_bf16 per K-step per wave.
// K panels (A: 128x32, W: 128x32) are DMA'd into double-buffered LDS by the
// Tensor Data Mover; wave 0 issues the next panel's descriptors while all
// waves compute on the current one (s_wait_tensorcnt 2 = current complete).
// VTRANS: scatter the output per-head transposed (for the V projection):
//   element (m = b*S+s, n = h*64+d) -> out[((b*16+h)*64+d)*1024 + s]
// ---------------------------------------------------------------------------
template <bool BIAS, bool RELU, bool RESID, bool OUTBF, bool VTRANS>
__global__ __launch_bounds__(128)
void gemm_bf16_wmma(const bf16_t* __restrict__ A, const bf16_t* __restrict__ W,
                    const float* __restrict__ bias, const float* __restrict__ resid,
                    float* __restrict__ outF, bf16_t* __restrict__ outB,
                    int N, int K) {
  __shared__ bf16_t ldsA[2][128 * 32];   // 8 KB per buffer
  __shared__ bf16_t ldsB[2][128 * 32];

  const int wave = threadIdx.x >> 5;
  const int lane = threadIdx.x & 31;
  const int m0b = blockIdx.y * 128;
  const int n0b = blockIdx.x * 128;
  const int mW  = (wave >> 1) * 64;      // wave tile inside block tile
  const int nW  = (wave & 1) * 64;

  const bf16_t* Abase = A + (size_t)m0b * K;
  const bf16_t* Wbase = W + (size_t)n0b * K;
  const uint32_t offA[2] = { (uint32_t)(uintptr_t)&ldsA[0][0],
                             (uint32_t)(uintptr_t)&ldsA[1][0] };
  const uint32_t offB[2] = { (uint32_t)(uintptr_t)&ldsB[0][0],
                             (uint32_t)(uintptr_t)&ldsB[1][0] };

  v8f acc[4][4] = {};
  const int nk = K >> 5;

  if (wave == 0) {                        // prologue: DMA first panels
    tdm_load_panel(offA[0], Abase, K, 128);
    tdm_load_panel(offB[0], Wbase, K, 128);
  }

  int cur = 0;
  for (int s = 0; s < nk; ++s) {
    if (wave == 0) {
      if (s + 1 < nk) {                   // prefetch next panels, then wait
        tdm_load_panel(offA[cur ^ 1], Abase + (s + 1) * 32, K, 128);
        tdm_load_panel(offB[cur ^ 1], Wbase + (s + 1) * 32, K, 128);
        __builtin_amdgcn_s_wait_tensorcnt(2);   // current 2 descriptors done
      } else {
        __builtin_amdgcn_s_wait_tensorcnt(0);
      }
    }
    __syncthreads();                      // publish panel block-wide

    const bf16_t* pa = &ldsA[cur][0];
    const bf16_t* pb = &ldsB[cur][0];
    v16bf a[4], b[4];
#pragma unroll
    for (int i = 0; i < 4; ++i) a[i] = load_fragA(pa + (mW + 16 * i) * 32, 32);
#pragma unroll
    for (int j = 0; j < 4; ++j) b[j] = load_fragB(pb + (nW + 16 * j) * 32, 32);
#pragma unroll
    for (int i = 0; i < 4; ++i)
#pragma unroll
      for (int j = 0; j < 4; ++j)
        acc[i][j] = wmma_bf16(a[i], b[j], acc[i][j]);

    __syncthreads();                      // done reading before overwrite
    cur ^= 1;
  }

  const int colL   = lane & 15;
  const int rowSel = (lane >> 4) << 3;
#pragma unroll
  for (int i = 0; i < 4; ++i) {
#pragma unroll
    for (int j = 0; j < 4; ++j) {
#pragma unroll
      for (int r = 0; r < 8; ++r) {
        int row = m0b + mW + 16 * i + rowSel + r;
        int col = n0b + nW + 16 * j + colL;
        float v = acc[i][j][r];
        if (BIAS)  v += bias[col];
        if (RELU)  v = fmaxf(v, 0.0f);
        if (RESID) v += resid[(size_t)row * N + col];
        if (OUTBF) {
          if (VTRANS) {
            int bb = row >> 10, ss = row & 1023;     // S = 1024
            int hh = col >> 6,  dd = col & 63;       // hd = 64
            outB[((size_t)((bb * NHEADS + hh) << 6) + dd) * SEQ + ss] = (bf16_t)v;
          } else {
            outB[(size_t)row * N + col] = (bf16_t)v;
          }
        } else {
          outF[(size_t)row * N + col] = v;
        }
      }
    }
  }
}

// ---------------------------------------------------------------------------
// Causal flash attention. One wave per (batch, head, 16-query-row) tile.
// Q,K: bf16 [B,S,D] (head slice ld = D).  VT: bf16 [B,H,hd,S].
// Online softmax over 32-key blocks; Q*K^T and P*V both via WMMA.
// ---------------------------------------------------------------------------
__global__ __launch_bounds__(128)
void attn_kernel(const bf16_t* __restrict__ Q, const bf16_t* __restrict__ Kb,
                 const bf16_t* __restrict__ VT, bf16_t* __restrict__ O) {
  __shared__ bf16_t ldsP[4][16 * 32];           // per-wave P staging tile

  const int wave = threadIdx.x >> 5;
  const int lane = threadIdx.x & 31;
  const int t    = blockIdx.x * 4 + wave;       // 0..4095 tiles
  const int b    = t >> 10;                     // / (NHEADS * S/16)
  const int head = (t >> 6) & (NHEADS - 1);
  const int qt   = t & 63;
  const int qr0  = qt * 16;

  const int colL   = lane & 15;
  const int rowSel = (lane >> 4) << 3;

  const bf16_t* qp = Q + ((size_t)(b * SEQ + qr0) * D_MODEL + head * HDIM);
  const v16bf aq0 = load_fragA(qp, D_MODEL);          // d 0..31
  const v16bf aq1 = load_fragA(qp + 32, D_MODEL);     // d 32..63

  float mR[8], lR[8];
  v8f o0 = {}, o1 = {}, o2 = {}, o3 = {};
#pragma unroll
  for (int r = 0; r < 8; ++r) { mR[r] = -1e30f; lR[r] = 0.0f; }

  bf16_t* myP = ldsP[wave];
  const int kbMax = (qr0 + 15) >> 5;

  for (int kb = 0; kb <= kbMax; ++kb) {
    const int kr0 = kb * 32;
    const bf16_t* kp = Kb + ((size_t)(b * SEQ + kr0) * D_MODEL + head * HDIM);
    // scores: Q(16x64) x K^T(64x32) -> two 16x16 tiles
    v16bf k00 = load_fragB(kp, D_MODEL);                       // keys kr0..+15, d0..31
    v16bf k01 = load_fragB(kp + 32, D_MODEL);                  // keys kr0..+15, d32..63
    v16bf k10 = load_fragB(kp + 16 * D_MODEL, D_MODEL);        // keys +16..+31
    v16bf k11 = load_fragB(kp + 16 * D_MODEL + 32, D_MODEL);
    v8f s0 = {}, s1 = {};
    s0 = wmma_bf16(aq0, k00, s0); s0 = wmma_bf16(aq1, k01, s0);
    s1 = wmma_bf16(aq0, k10, s1); s1 = wmma_bf16(aq1, k11, s1);

    float p0[8], p1[8];
#pragma unroll
    for (int r = 0; r < 8; ++r) {
      const int rowg = qr0 + rowSel + r;
      const int c0 = kr0 + colL;
      const int c1 = c0 + 16;
      float x0 = s0[r] * 0.125f; if (c0 > rowg) x0 = -1e30f;   // 1/sqrt(64)
      float x1 = s1[r] * 0.125f; if (c1 > rowg) x1 = -1e30f;
      float rm = fmaxf(x0, x1);
      rm = fmaxf(rm, __shfl_xor(rm, 1));
      rm = fmaxf(rm, __shfl_xor(rm, 2));
      rm = fmaxf(rm, __shfl_xor(rm, 4));
      rm = fmaxf(rm, __shfl_xor(rm, 8));      // row-wise over the 16-lane half
      const float mNew = fmaxf(mR[r], rm);
      const float fac  = __expf(mR[r] - mNew);
      const float e0 = __expf(x0 - mNew);
      const float e1 = __expf(x1 - mNew);
      float rs = e0 + e1;
      rs += __shfl_xor(rs, 1);
      rs += __shfl_xor(rs, 2);
      rs += __shfl_xor(rs, 4);
      rs += __shfl_xor(rs, 8);
      lR[r] = lR[r] * fac + rs;
      mR[r] = mNew;
      o0[r] *= fac; o1[r] *= fac; o2[r] *= fac; o3[r] *= fac;
      p0[r] = e0; p1[r] = e1;
    }

    // C-layout -> A-layout for P via per-wave LDS tile (row-major 16x32)
#pragma unroll
    for (int r = 0; r < 8; ++r) {
      const int row = rowSel + r;
      myP[row * 32 + colL]      = (bf16_t)p0[r];
      myP[row * 32 + 16 + colL] = (bf16_t)p1[r];
    }
    __builtin_amdgcn_wave_barrier();  // keep DS order; same-wave DS is in-order
    const v16bf ap = load_fragA(myP, 32);

    // o += P(16x32) x V(32x64): VT is [d][s] row-major -> contiguous B frags
    const bf16_t* vp = VT + ((size_t)((b * NHEADS + head) * HDIM) * SEQ + kr0);
    v16bf vb0 = load_fragB(vp + 0 * 16 * SEQ, SEQ);
    v16bf vb1 = load_fragB(vp + 1 * 16 * SEQ, SEQ);
    v16bf vb2 = load_fragB(vp + 2 * 16 * SEQ, SEQ);
    v16bf vb3 = load_fragB(vp + 3 * 16 * SEQ, SEQ);
    o0 = wmma_bf16(ap, vb0, o0);
    o1 = wmma_bf16(ap, vb1, o1);
    o2 = wmma_bf16(ap, vb2, o2);
    o3 = wmma_bf16(ap, vb3, o3);
  }

  // normalize and store attention output as bf16 [B,S,D]
#pragma unroll
  for (int r = 0; r < 8; ++r) {
    const float inv = 1.0f / lR[r];
    const size_t base =
        (size_t)(b * SEQ + qr0 + rowSel + r) * D_MODEL + head * HDIM + colL;
    O[base + 0]  = (bf16_t)(o0[r] * inv);
    O[base + 16] = (bf16_t)(o1[r] * inv);
    O[base + 32] = (bf16_t)(o2[r] * inv);
    O[base + 48] = (bf16_t)(o3[r] * inv);
  }
}

// ---------------------------------------------------------------------------
// Host-side orchestration
// ---------------------------------------------------------------------------
static inline void* ws_alloc(char*& p, size_t bytes) {
  void* r = (void*)p;
  p += (bytes + 255) & ~(size_t)255;
  return r;
}

extern "C" void kernel_launch(void* const* d_in, const int* in_sizes, int n_in,
                              void* d_out, int out_size, void* d_ws, size_t ws_size,
                              hipStream_t stream) {
  (void)in_sizes; (void)n_in; (void)out_size; (void)ws_size;

  const int*   x    = (const int*)d_in[0];
  const float* embW = (const float*)d_in[1];
  const float* qW   = (const float*)d_in[2];
  const float* kW   = (const float*)d_in[3];
  const float* vW   = (const float*)d_in[4];
  const float* oW   = (const float*)d_in[5];
  const float* ln0g = (const float*)d_in[6];
  const float* ln0b = (const float*)d_in[7];
  const float* ln1g = (const float*)d_in[8];
  const float* ln1b = (const float*)d_in[9];
  const float* W1   = (const float*)d_in[10];
  const float* b1   = (const float*)d_in[11];
  const float* W2   = (const float*)d_in[12];
  const float* b2   = (const float*)d_in[13];
  const float* outW = (const float*)d_in[14];
  const float* outb = (const float*)d_in[15];

  char* wp = (char*)d_ws;
  float*  h     = (float*) ws_alloc(wp, (size_t)NTOK * D_MODEL * 4);
  bf16_t* aB    = (bf16_t*)ws_alloc(wp, (size_t)NTOK * D_MODEL * 2);
  bf16_t* qB    = (bf16_t*)ws_alloc(wp, (size_t)NTOK * D_MODEL * 2);
  bf16_t* kB    = (bf16_t*)ws_alloc(wp, (size_t)NTOK * D_MODEL * 2);
  bf16_t* vTB   = (bf16_t*)ws_alloc(wp, (size_t)NTOK * D_MODEL * 2);
  bf16_t* oB    = (bf16_t*)ws_alloc(wp, (size_t)NTOK * D_MODEL * 2);
  bf16_t* ff1B  = (bf16_t*)ws_alloc(wp, (size_t)NTOK * HFF * 2);
  bf16_t* WqB   = (bf16_t*)ws_alloc(wp, (size_t)D_MODEL * D_MODEL * 2);
  bf16_t* WkB   = (bf16_t*)ws_alloc(wp, (size_t)D_MODEL * D_MODEL * 2);
  bf16_t* WvB   = (bf16_t*)ws_alloc(wp, (size_t)D_MODEL * D_MODEL * 2);
  bf16_t* WoB   = (bf16_t*)ws_alloc(wp, (size_t)D_MODEL * D_MODEL * 2);
  bf16_t* W1B   = (bf16_t*)ws_alloc(wp, (size_t)HFF * D_MODEL * 2);
  bf16_t* W2B   = (bf16_t*)ws_alloc(wp, (size_t)D_MODEL * HFF * 2);
  bf16_t* WoutB = (bf16_t*)ws_alloc(wp, (size_t)VOCAB * D_MODEL * 2);

  // 1) weight conversion f32 -> bf16
  {
    const int nDD = D_MODEL * D_MODEL;
    const int nHD = HFF * D_MODEL;
    const int nVD = VOCAB * D_MODEL;
    cvt_f32_bf16<<<(nDD + 255) / 256, 256, 0, stream>>>(qW,   WqB,   nDD);
    cvt_f32_bf16<<<(nDD + 255) / 256, 256, 0, stream>>>(kW,   WkB,   nDD);
    cvt_f32_bf16<<<(nDD + 255) / 256, 256, 0, stream>>>(vW,   WvB,   nDD);
    cvt_f32_bf16<<<(nDD + 255) / 256, 256, 0, stream>>>(oW,   WoB,   nDD);
    cvt_f32_bf16<<<(nHD + 255) / 256, 256, 0, stream>>>(W1,   W1B,   nHD);
    cvt_f32_bf16<<<(nHD + 255) / 256, 256, 0, stream>>>(W2,   W2B,   nHD);
    cvt_f32_bf16<<<(nVD + 255) / 256, 256, 0, stream>>>(outW, WoutB, nVD);
  }

  // 2) embedding + positional encoding
  embed_pe_kernel<<<NTOK, 256, 0, stream>>>(x, embW, h);

  const dim3 gD(D_MODEL / 128, NTOK / 128);   // (8, 32)
  const dim3 gF(HFF / 128,     NTOK / 128);   // (32, 32)
  const dim3 gV(VOCAB / 128,   NTOK / 128);   // (250, 32)

  // 3) transformer blocks (weights shared)
  for (int blk = 0; blk < 4; ++blk) {
    add_ln_kernel<<<NTOK, 256, 0, stream>>>(h, ln0g, ln0b, aB);
    gemm_bf16_wmma<false, false, false, true, false>
        <<<gD, 128, 0, stream>>>(aB, WqB, nullptr, nullptr, nullptr, qB,
                                 D_MODEL, D_MODEL);
    gemm_bf16_wmma<false, false, false, true, false>
        <<<gD, 128, 0, stream>>>(aB, WkB, nullptr, nullptr, nullptr, kB,
                                 D_MODEL, D_MODEL);
    gemm_bf16_wmma<false, false, false, true, true>
        <<<gD, 128, 0, stream>>>(aB, WvB, nullptr, nullptr, nullptr, vTB,
                                 D_MODEL, D_MODEL);
    attn_kernel<<<NTOK / 4, 128, 0, stream>>>(qB, kB, vTB, oB);
    gemm_bf16_wmma<false, false, true, false, false>
        <<<gD, 128, 0, stream>>>(oB, WoB, nullptr, h, h, nullptr,
                                 D_MODEL, D_MODEL);
    add_ln_kernel<<<NTOK, 256, 0, stream>>>(h, ln1g, ln1b, aB);
    gemm_bf16_wmma<true, true, false, true, false>
        <<<gF, 128, 0, stream>>>(aB, W1B, b1, nullptr, nullptr, ff1B,
                                 HFF, D_MODEL);
    gemm_bf16_wmma<true, false, true, false, false>
        <<<gD, 128, 0, stream>>>(ff1B, W2B, b2, h, h, nullptr,
                                 D_MODEL, HFF);
  }

  // 4) logits = h_bf16 @ outW^T + out_b  -> d_out (fp32)
  cvt_f32_bf16<<<(NTOK * D_MODEL + 255) / 256, 256, 0, stream>>>(
      h, aB, NTOK * D_MODEL);
  gemm_bf16_wmma<true, false, false, false, false>
      <<<gV, 128, 0, stream>>>(aB, WoutB, outb, nullptr, (float*)d_out, nullptr,
                               VOCAB, D_MODEL);
}
